// CostVolumeV2_14877766714073
// MI455X (gfx1250) — matrix operand order repaired and checked by reading, not compiled, not run on hardware
//
#include <hip/hip_runtime.h>
#include <stdint.h>

typedef __attribute__((ext_vector_type(16))) _Float16 v16h;
typedef __attribute__((ext_vector_type(8)))  float    v8f;
typedef __attribute__((ext_vector_type(4)))  float    f4;

#define BB 8
#define HH 128
#define WW 128
#define CC 192
#define ND 9            // 2*SEARCH_RANGE+1
#define NOUT 81
#define COLS 144        // LDS columns: global col -4 .. 139 (halo, zero-padded)
#define CSTR 196        // dword stride per column (192 data + 4 pad -> conflict-free frags)
#define BUFDW (COLS*CSTR)
#define THREADS 256

extern __shared__ float smem[];

__device__ __forceinline__ void async_ld_b128(uint32_t lds, const float* g) {
  asm volatile("global_load_async_to_lds_b128 %0, %1, off"
               :: "v"(lds), "v"((uint64_t)(uintptr_t)g) : "memory");
}
__device__ __forceinline__ void wait_async0() {
  asm volatile("s_wait_asynccnt 0" ::: "memory");
}

__device__ __forceinline__ f4 ld4(const float* p) { return *(const f4*)p; }

__device__ __forceinline__ v16h cvt16(f4 a, f4 b, f4 c, f4 d) {
  v16h r;
  r[0]=(_Float16)a[0];  r[1]=(_Float16)a[1];  r[2]=(_Float16)a[2];  r[3]=(_Float16)a[3];
  r[4]=(_Float16)b[0];  r[5]=(_Float16)b[1];  r[6]=(_Float16)b[2];  r[7]=(_Float16)b[3];
  r[8]=(_Float16)c[0];  r[9]=(_Float16)c[1];  r[10]=(_Float16)c[2]; r[11]=(_Float16)c[3];
  r[12]=(_Float16)d[0]; r[13]=(_Float16)d[1]; r[14]=(_Float16)d[2]; r[15]=(_Float16)d[3];
  return r;
}

// B fragment (32x16 f16): lane = (n, half). K = e + 16*half -> 16 consecutive f32 in LDS.
__device__ __forceinline__ v16h load_bfrag(const float* buf, int ldscol, int kb, int hb) {
  const float* q = buf + ldscol*CSTR + kb*32 + hb*16;
  return cvt16(ld4(q), ld4(q+4), ld4(q+8), ld4(q+12));
}

// Stage one nxt row (f32) into LDS cols 4..131 via async global->LDS DMA.
// Out-of-range rows are zero-filled (matches jnp.pad).
__device__ __forceinline__ void stage_row(float* buf, const float* nxt, int b, int row, int tid) {
  if (row >= 0 && row < HH) {
    const float* src = nxt + ((size_t)(b*HH + row))*WW*CC;
    uint32_t base = (uint32_t)(uintptr_t)buf;
    for (int i = tid; i < WW*(CC/4); i += THREADS) {
      int col = i / (CC/4);
      int ch  = i % (CC/4);
      async_ld_b128(base + (uint32_t)(((col+4)*CSTR + ch*4)*4),
                    src + (size_t)col*CC + ch*4);
    }
  } else {
    f4 z = {0.f,0.f,0.f,0.f};
    for (int i = tid; i < WW*(CC/4); i += THREADS) {
      int col = i/(CC/4), ch = i%(CC/4);
      *(f4*)(buf + (col+4)*CSTR + ch*4) = z;
    }
  }
}

__global__ void __launch_bounds__(THREADS)
costvol_wmma_kernel(const float* __restrict__ prv, const float* __restrict__ nxt,
                    float* __restrict__ out)
{
  const int tid  = threadIdx.x;
  const int wave = tid >> 5;
  const int lane = tid & 31;
  const int n    = lane & 15;   // A-row m, B/D column n
  const int half = lane >> 4;   // lane-half for A/B K offsets and D M offset
  const int w0   = wave * 16;   // this wave's 16-pixel M tile
  const int bh   = blockIdx.x;
  const int b    = bh / HH, h = bh % HH;

  float* buf0 = smem;
  float* buf1 = smem + BUFDW;

  // Zero halo columns (global cols -4..-1 and 128..139) of both buffers once.
  {
    f4 z = {0.f,0.f,0.f,0.f};
    const int per = 16*(CSTR/4);
    for (int i = tid; i < 2*per; i += THREADS) {
      int bi = i / per, r = i % per;
      int j = r / (CSTR/4), c4 = r % (CSTR/4);
      int col = (j < 4) ? j : (j + 128);
      float* buf = bi ? buf1 : buf0;
      *(f4*)(buf + col*CSTR + c4*4) = z;
    }
  }

  // Kick off async staging of the first nxt row (dy=0 -> h-4).
  stage_row(buf0, nxt, b, h - 4, tid);

  // A fragments (16x32 f16 per k-block): lane m=n, K = e + 16*half.
  v16h afrag[6];
  {
    const float* ppix = prv + (((size_t)(b*HH + h))*WW + (size_t)(w0 + n)) * CC;
#pragma unroll
    for (int kb = 0; kb < 6; ++kb) {
      const float* p = ppix + kb*32 + half*16;
      afrag[kb] = cvt16(ld4(p), ld4(p+4), ld4(p+8), ld4(p+12));
    }
  }

  wait_async0();
  __syncthreads();

  const float inv = 1.0f / (float)CC;
  const size_t rowbase = ((size_t)(b*HH) + (size_t)h) * (size_t)WW;

  for (int dy = 0; dy < ND; ++dy) {
    float* cur = (dy & 1) ? buf1 : buf0;
    float* nb  = (dy & 1) ? buf0 : buf1;
    if (dy < ND-1) stage_row(nb, nxt, b, h + dy + 1 - 4, tid);  // overlap with compute

    // Two banded 16x16 correlation tiles: B cols at global offsets -4 and +8.
    v8f acc0 = {0,0,0,0,0,0,0,0};
    v8f acc1 = {0,0,0,0,0,0,0,0};
#pragma unroll
    for (int kb = 0; kb < 6; ++kb) {
      v16h b0 = load_bfrag(cur, w0 + n,      kb, half);   // LDS col = global col + 4
      v16h b1 = load_bfrag(cur, w0 + 12 + n, kb, half);
      acc0 = __builtin_amdgcn_wmma_f32_16x16x32_f16(false, afrag[kb], false, b0,
                                                    (short)0, acc0, false, false);
      acc1 = __builtin_amdgcn_wmma_f32_16x16x32_f16(false, afrag[kb], false, b1,
                                                    (short)0, acc1, false, false);
    }

    // Extract the dx band, apply mean + leaky ReLU, scatter-store.
    // acc0 holds dx = n-4-i (used when i+dx <= 11); acc1 holds dx = 8+n-i (i+dx >= 12 <=> n >= 4).
#pragma unroll
    for (int r = 0; r < 8; ++r) {
      int i = r + 8*half;   // D-matrix M index
      int dx0 = n - 4 - i;
      if (dx0 >= -4 && dx0 <= 4) {
        float v = acc0[r] * inv;
        v = (v > 0.f) ? v : 0.1f * v;
        out[(rowbase + (size_t)(w0 + i))*NOUT + dy*ND + (dx0 + 4)] = v;
      }
      int dx1 = 8 + n - i;
      if (n >= 4 && dx1 >= -4 && dx1 <= 4) {
        float v = acc1[r] * inv;
        v = (v > 0.f) ? v : 0.1f * v;
        out[(rowbase + (size_t)(w0 + i))*NOUT + dy*ND + (dx1 + 4)] = v;
      }
    }

    wait_async0();     // next buffer fully landed in LDS (this wave's transfers)
    __syncthreads();   // publish to all waves; also fences LDS reads of 'cur'
  }
}

extern "C" void kernel_launch(void* const* d_in, const int* in_sizes, int n_in,
                              void* d_out, int out_size, void* d_ws, size_t ws_size,
                              hipStream_t stream) {
  (void)in_sizes; (void)n_in; (void)d_ws; (void)ws_size; (void)out_size;
  const float* prv = (const float*)d_in[0];
  const float* nxt = (const float*)d_in[1];
  float* out = (float*)d_out;

  const size_t shmem = (size_t)2 * BUFDW * sizeof(float);   // ~226 KB of the 320 KB WGP LDS
  (void)hipFuncSetAttribute((const void*)costvol_wmma_kernel,
                            hipFuncAttributeMaxDynamicSharedMemorySize, (int)shmem);
  costvol_wmma_kernel<<<dim3(BB*HH), dim3(THREADS), shmem, stream>>>(prv, nxt, out);
}
